// LlamaModel_27702539059454
// MI455X (gfx1250) — matrix operand order repaired and checked by reading, not compiled, not run on hardware
//
#include <hip/hip_runtime.h>
#include <stdint.h>

// ---------------- model constants ----------------
#define H      2048
#define S_LEN  1024
#define BATCH  2
#define NTOK   (BATCH * S_LEN)           // 2048 tokens
#define NHEAD  32
#define NKV    8
#define HD     64
#define QKV_N  ((NHEAD + 2 * NKV) * HD)  // 3072
#define FFI    5632
#define NLAYER 4
#define EPSF   1e-5f

// ---------------- WMMA types ----------------
typedef __attribute__((ext_vector_type(8)))  float  v8f;
typedef __attribute__((ext_vector_type(16))) __bf16 v16bf;

struct U128 { unsigned int x, y, z, w; };   // trivial 16B chunk
union  FragBF { v16bf v; U128 u[2]; };      // 32B: one WMMA A/B fragment

// ---- CDNA5 async global->LDS copy (16B per lane), tracked by ASYNCcnt ----
__device__ __forceinline__ void async_copy16(const void* gp, void* lp) {
    // generic LDS pointers carry the LDS byte offset in the low 32 bits
    // (LDS aperture check is on addr[63:32]); global low 64 bits = VA.
    unsigned loff = (unsigned)(uintptr_t)lp;
    unsigned long long ga = (unsigned long long)(uintptr_t)gp;
    asm volatile("global_load_async_to_lds_b128 %0, %1, off"
                 :: "v"(loff), "v"(ga)
                 : "memory");
}
#define WAIT_ASYNC_4() asm volatile("s_wait_asynccnt 4" ::: "memory")
#define WAIT_ASYNC_0() asm volatile("s_wait_asynccnt 0" ::: "memory")

// ---------------- 1. embedding gather ----------------
__global__ __launch_bounds__(256) void embed_kernel(const int* __restrict__ ids,
                                                    const float* __restrict__ w,
                                                    float* __restrict__ out) {
    int idx = blockIdx.x * 256 + threadIdx.x;     // NTOK*H threads
    int tok = idx >> 11;                          // /H
    int c   = idx & (H - 1);
    out[idx] = w[(size_t)ids[tok] * H + c];
}

// ---------------- 2. weight transpose + fp32->bf16 (once per launch) ----------------
// src: [K,N] fp32 row-major  ->  dst: [N,K] bf16 row-major
__global__ __launch_bounds__(256) void transpose_bf16_kernel(const float* __restrict__ src,
                                                             __bf16* __restrict__ dst,
                                                             int K, int N) {
    __shared__ float tile[32][33];
    int n0 = blockIdx.x * 32;
    int k0 = blockIdx.y * 32;
    int c  = threadIdx.x & 31;
    int r0 = (threadIdx.x >> 5) * 4;
#pragma unroll
    for (int i = 0; i < 4; ++i) {
        int r = r0 + i;
        tile[r][c] = src[(size_t)(k0 + r) * N + n0 + c];
    }
    __syncthreads();
#pragma unroll
    for (int i = 0; i < 4; ++i) {
        int r = r0 + i;   // row within n-tile
        dst[(size_t)(n0 + r) * K + k0 + c] = (__bf16)tile[c][r];
    }
}

// ---------------- 3. fused residual-add + RMSNorm ----------------
// r = first ? h : h + resid ; resid = r ; x = r * rsqrt(mean(r^2)+eps) * w
// writes bf16 (GEMM operand) or fp32 (final output)
__global__ __launch_bounds__(256) void add_rms_kernel(const float* __restrict__ h,
                                                      float* __restrict__ resid,
                                                      const float* __restrict__ w,
                                                      __bf16* __restrict__ xbf,
                                                      float* __restrict__ xf,
                                                      int first) {
    __shared__ float red[256];
    int row = blockIdx.x;
    int tid = threadIdx.x;
    const float* hr = h     + (size_t)row * H;
    float*       rr = resid + (size_t)row * H;

    float v[8];
    float ss = 0.0f;
#pragma unroll
    for (int i = 0; i < 8; ++i) {
        int c = tid + i * 256;
        float t = hr[c];
        if (!first) t += rr[c];
        v[i] = t;
        ss += t * t;
    }
    red[tid] = ss;
    __syncthreads();
    for (int off = 128; off > 0; off >>= 1) {
        if (tid < off) red[tid] += red[tid + off];
        __syncthreads();
    }
    float rs = rsqrtf(red[0] * (1.0f / (float)H) + EPSF);
#pragma unroll
    for (int i = 0; i < 8; ++i) {
        int c = tid + i * 256;
        rr[c] = v[i];
        float o = v[i] * rs * w[c];
        if (xf) xf[(size_t)row * H + c] = o;
        else    xbf[(size_t)row * H + c] = (__bf16)o;
    }
}

// ---------------- 4. bf16 WMMA GEMM with async double-buffered LDS staging ----------------
// C[M,N](fp32) = A[M,K](bf16, row-major) x Bt[N,K](bf16, pre-transposed)
// 128x128 tile, BK=32, 8 waves (4M x 2N), each wave 32x64 = 2x4 WMMA 16x16x32.
#define BM 128
#define BN 128
#define BK 32
#define LDSK 40   // padded K-stride in bf16 elems; 80B rows keep 16B align, dodge bank conflicts

__global__ __launch_bounds__(256) void gemm_bf16_wmma(const __bf16* __restrict__ A,
                                                      const __bf16* __restrict__ Bt,
                                                      float* __restrict__ C,
                                                      int M, int N, int K) {
    __shared__ unsigned short As[2][BM * LDSK];
    __shared__ unsigned short Bs[2][BN * LDSK];

    const int tid  = threadIdx.x;
    const int lane = tid & 31;
    const int wave = tid >> 5;
    const int waveM = (wave >> 1) * 32;        // 0,32,64,96
    const int waveN = (wave & 1) * 64;         // 0,64
    const int kh = lane >> 4;                  // K-half select (16-bit A/B layouts)
    const int ml = lane & 15;                  // row (A) / col (B,C)
    const int blockM = blockIdx.y * BM;
    const int blockN = blockIdx.x * BN;

    // async staging: each tile = 128 rows x 64B = 512 x 16B chunks; thread owns 2 chunks
    const int c0   = tid * 2;
    const int row0 = c0 >> 2, ch0 = (c0 & 3) * 8;        // chunk offset in bf16 elems
    const int c1   = c0 + 1;
    const int row1 = c1 >> 2, ch1 = (c1 & 3) * 8;

    v8f acc[2][4];
#pragma unroll
    for (int mi = 0; mi < 2; ++mi)
#pragma unroll
        for (int ni = 0; ni < 4; ++ni)
#pragma unroll
            for (int e = 0; e < 8; ++e) acc[mi][ni][e] = 0.0f;

    // prologue: stage tile 0 into buffer 0
    {
        async_copy16(A  + (size_t)(blockM + row0) * K + ch0, &As[0][row0 * LDSK + ch0]);
        async_copy16(A  + (size_t)(blockM + row1) * K + ch1, &As[0][row1 * LDSK + ch1]);
        async_copy16(Bt + (size_t)(blockN + row0) * K + ch0, &Bs[0][row0 * LDSK + ch0]);
        async_copy16(Bt + (size_t)(blockN + row1) * K + ch1, &Bs[0][row1 * LDSK + ch1]);
    }

    const int nTiles = K / BK;
    for (int t = 0; t < nTiles; ++t) {
        const int cur = t & 1;
        const bool hasNext = (t + 1) < nTiles;
        if (hasNext) {
            const int k0 = (t + 1) * BK;
            const int nb = cur ^ 1;
            async_copy16(A  + (size_t)(blockM + row0) * K + k0 + ch0, &As[nb][row0 * LDSK + ch0]);
            async_copy16(A  + (size_t)(blockM + row1) * K + k0 + ch1, &As[nb][row1 * LDSK + ch1]);
            async_copy16(Bt + (size_t)(blockN + row0) * K + k0 + ch0, &Bs[nb][row0 * LDSK + ch0]);
            async_copy16(Bt + (size_t)(blockN + row1) * K + k0 + ch1, &Bs[nb][row1 * LDSK + ch1]);
            WAIT_ASYNC_4();          // async loads retire in order -> tile t's 4 copies done
        } else {
            WAIT_ASYNC_0();
        }
        __syncthreads();             // all waves' copies for tile t visible

        // ---- fragments (documented 16-bit A/B VGPR layouts) ----
        FragBF a[2], b[4];
#pragma unroll
        for (int mi = 0; mi < 2; ++mi) {
            const unsigned short* rp = &As[cur][(waveM + mi * 16 + ml) * LDSK];
            a[mi].u[0] = *(const U128*)(rp + kh * 8);        // K = kh*8 .. +7
            a[mi].u[1] = *(const U128*)(rp + 16 + kh * 8);   // K = 16+kh*8 .. +7
        }
#pragma unroll
        for (int ni = 0; ni < 4; ++ni) {
            const unsigned short* cp = &Bs[cur][(waveN + ni * 16 + ml) * LDSK];
            b[ni].u[0] = *(const U128*)(cp + kh * 8);        // K = kh*8 .. +7
            b[ni].u[1] = *(const U128*)(cp + 16 + kh * 8);   // K = 16+kh*8 .. +7
        }
#pragma unroll
        for (int mi = 0; mi < 2; ++mi)
#pragma unroll
            for (int ni = 0; ni < 4; ++ni)
                acc[mi][ni] = __builtin_amdgcn_wmma_f32_16x16x32_bf16(
                    false, a[mi].v, false, b[ni].v, (short)0, acc[mi][ni], false, false);
        __syncthreads();             // done reading buffer before it is re-staged
    }

    // ---- store: 32-bit C/D layout: lane ml = N, VGPR r -> M = r + kh*8 ----
#pragma unroll
    for (int mi = 0; mi < 2; ++mi)
#pragma unroll
        for (int ni = 0; ni < 4; ++ni) {
            int col = blockN + waveN + ni * 16 + ml;
#pragma unroll
            for (int r = 0; r < 8; ++r) {
                int row = blockM + waveM + mi * 16 + kh * 8 + r;
                C[(size_t)row * N + col] = acc[mi][ni][r];
            }
        }
}

// ---------------- 5. RoPE on q and k heads of qkv buffer ----------------
__global__ __launch_bounds__(256) void rope_kernel(float* __restrict__ qkv,
                                                   const int* __restrict__ pos) {
    int idx  = blockIdx.x * 256 + threadIdx.x;    // NTOK * 40 heads * 32 dims
    int d    = idx & 31;
    int head = (idx >> 5) % (NHEAD + NKV);
    int tok  = idx / ((NHEAD + NKV) * 32);
    float* base = qkv + (size_t)tok * QKV_N +
                  (head < NHEAD ? head * HD : NHEAD * HD + (head - NHEAD) * HD);
    float p   = (float)pos[tok];
    float inv = __powf(10000.0f, -(float)d * (1.0f / 32.0f));
    float sn, cs;
    __sincosf(p * inv, &sn, &cs);
    float x1 = base[d], x2 = base[d + 32];
    base[d]      = x1 * cs - x2 * sn;
    base[d + 32] = x2 * cs + x1 * sn;
}

// ---------------- 6. causal attention, one thread per query row (online softmax) ----------------
__global__ __launch_bounds__(256) void attn_kernel(const float* __restrict__ qkv,
                                                   __bf16* __restrict__ out) {
    int idx = blockIdx.x * 256 + threadIdx.x;     // B*NHEAD*S_LEN
    int s   = idx % S_LEN;
    int qh  = (idx / S_LEN) % NHEAD;
    int b   = idx / (S_LEN * NHEAD);
    int kvh = qh >> 2;                            // GQA group = 4
    const float scale = 0.125f;                   // HD^-0.5

    const float4* q4 = (const float4*)(qkv + (size_t)(b * S_LEN + s) * QKV_N + qh * HD);
    float4 q[16];
#pragma unroll
    for (int i = 0; i < 16; ++i) {
        float4 t = q4[i];
        t.x *= scale; t.y *= scale; t.z *= scale; t.w *= scale;
        q[i] = t;
    }
    float m = -1e30f, l = 0.0f;
    float4 acc[16];
#pragma unroll
    for (int i = 0; i < 16; ++i) acc[i] = make_float4(0.f, 0.f, 0.f, 0.f);

    for (int t = 0; t <= s; ++t) {
        const float* rowp = qkv + (size_t)(b * S_LEN + t) * QKV_N;
        const float4* k4 = (const float4*)(rowp + NHEAD * HD + kvh * HD);
        float dot = 0.0f;
#pragma unroll
        for (int i = 0; i < 16; ++i) {
            float4 kv = k4[i];
            dot += q[i].x * kv.x + q[i].y * kv.y + q[i].z * kv.z + q[i].w * kv.w;
        }
        float mn   = fmaxf(m, dot);
        float corr = __expf(m - mn);
        float p    = __expf(dot - mn);
        l = l * corr + p;
        const float4* v4 = (const float4*)(rowp + (NHEAD + NKV) * HD + kvh * HD);
#pragma unroll
        for (int i = 0; i < 16; ++i) {
            float4 vv = v4[i];
            acc[i].x = acc[i].x * corr + p * vv.x;
            acc[i].y = acc[i].y * corr + p * vv.y;
            acc[i].z = acc[i].z * corr + p * vv.z;
            acc[i].w = acc[i].w * corr + p * vv.w;
        }
        m = mn;
    }
    float inv = 1.0f / l;
    __bf16* orow = out + (size_t)(b * S_LEN + s) * (NHEAD * HD) + qh * HD;
#pragma unroll
    for (int i = 0; i < 16; ++i) {
        orow[i * 4 + 0] = (__bf16)(acc[i].x * inv);
        orow[i * 4 + 1] = (__bf16)(acc[i].y * inv);
        orow[i * 4 + 2] = (__bf16)(acc[i].z * inv);
        orow[i * 4 + 3] = (__bf16)(acc[i].w * inv);
    }
}

// ---------------- 7. SiLU(gate) * up -> bf16 ----------------
__global__ __launch_bounds__(256) void silu_mul_kernel(const float* __restrict__ gu,
                                                       __bf16* __restrict__ act) {
    int idx = blockIdx.x * 256 + threadIdx.x;     // NTOK * FFI
    int row = idx / FFI;
    int c   = idx - row * FFI;
    float g = gu[(size_t)row * (2 * FFI) + c];
    float u = gu[(size_t)row * (2 * FFI) + FFI + c];
    float sg = g * (1.0f / (1.0f + __expf(-g)));
    act[idx] = (__bf16)(sg * u);
}

// ---------------- host-side orchestration ----------------
static inline char* ws_alloc(char*& p, size_t bytes) {
    char* r = p;
    p += (bytes + 255) & ~(size_t)255;
    return r;
}

extern "C" void kernel_launch(void* const* d_in, const int* in_sizes, int n_in,
                              void* d_out, int out_size, void* d_ws, size_t ws_size,
                              hipStream_t stream) {
    const int*   ids     = (const int*)d_in[0];
    const int*   pos     = (const int*)d_in[1];
    const float* embed_w = (const float*)d_in[2];
    const float* qkv_w   = (const float*)d_in[3];
    const float* o_w     = (const float*)d_in[4];
    const float* gu_w    = (const float*)d_in[5];
    const float* down_w  = (const float*)d_in[6];
    const float* ln1_w   = (const float*)d_in[7];
    const float* ln2_w   = (const float*)d_in[8];
    const float* norm_w  = (const float*)d_in[9];
    float* outp = (float*)d_out;

    // ---- workspace layout ----
    char* p = (char*)d_ws;
    float*  resid   = (float*) ws_alloc(p, (size_t)NTOK * H * 4);
    float*  hbuf    = (float*) ws_alloc(p, (size_t)NTOK * H * 4);
    float*  qkvbuf  = (float*) ws_alloc(p, (size_t)NTOK * QKV_N * 4);
    float*  gubuf   = (float*) ws_alloc(p, (size_t)NTOK * 2 * FFI * 4);
    __bf16* xbf     = (__bf16*)ws_alloc(p, (size_t)NTOK * H * 2);
    __bf16* attnbf  = (__bf16*)ws_alloc(p, (size_t)NTOK * H * 2);
    __bf16* actbf   = (__bf16*)ws_alloc(p, (size_t)NTOK * FFI * 2);
    __bf16* qkvT    = (__bf16*)ws_alloc(p, (size_t)NLAYER * H * QKV_N * 2);    // [N,K]=[3072,2048]
    __bf16* oT      = (__bf16*)ws_alloc(p, (size_t)NLAYER * H * H * 2);        // [2048,2048]
    __bf16* guT     = (__bf16*)ws_alloc(p, (size_t)NLAYER * H * 2 * FFI * 2);  // [11264,2048]
    __bf16* downT   = (__bf16*)ws_alloc(p, (size_t)NLAYER * FFI * H * 2);      // [2048,5632]

    dim3 blk(256);

    // ---- one-time (per launch) weight transpose + bf16 conversion ----
    for (int l = 0; l < NLAYER; ++l) {
        transpose_bf16_kernel<<<dim3(QKV_N / 32, H / 32), blk, 0, stream>>>(
            qkv_w + (size_t)l * H * QKV_N, qkvT + (size_t)l * H * QKV_N, H, QKV_N);
        transpose_bf16_kernel<<<dim3(H / 32, H / 32), blk, 0, stream>>>(
            o_w + (size_t)l * H * H, oT + (size_t)l * H * H, H, H);
        transpose_bf16_kernel<<<dim3((2 * FFI) / 32, H / 32), blk, 0, stream>>>(
            gu_w + (size_t)l * H * 2 * FFI, guT + (size_t)l * H * 2 * FFI, H, 2 * FFI);
        transpose_bf16_kernel<<<dim3(H / 32, FFI / 32), blk, 0, stream>>>(
            down_w + (size_t)l * FFI * H, downT + (size_t)l * FFI * H, FFI, H);
    }

    embed_kernel<<<(NTOK * H) / 256, blk, 0, stream>>>(ids, embed_w, hbuf);
    add_rms_kernel<<<NTOK, blk, 0, stream>>>(hbuf, resid, ln1_w, xbf, nullptr, 1);

    for (int l = 0; l < NLAYER; ++l) {
        if (l > 0)
            add_rms_kernel<<<NTOK, blk, 0, stream>>>(
                hbuf, resid, ln1_w + (size_t)l * H, xbf, nullptr, 0);

        // QKV projection: [NTOK,H] x [H,3072]
        gemm_bf16_wmma<<<dim3(QKV_N / BN, NTOK / BM), blk, 0, stream>>>(
            xbf, qkvT + (size_t)l * H * QKV_N, qkvbuf, NTOK, QKV_N, H);

        rope_kernel<<<(NTOK * (NHEAD + NKV) * 32) / 256, blk, 0, stream>>>(qkvbuf, pos);

        attn_kernel<<<(BATCH * NHEAD * S_LEN) / 256, blk, 0, stream>>>(qkvbuf, attnbf);

        // O projection: [NTOK,2048] x [2048,2048]
        gemm_bf16_wmma<<<dim3(H / BN, NTOK / BM), blk, 0, stream>>>(
            attnbf, oT + (size_t)l * H * H, hbuf, NTOK, H, H);

        add_rms_kernel<<<NTOK, blk, 0, stream>>>(
            hbuf, resid, ln2_w + (size_t)l * H, xbf, nullptr, 0);

        // gate+up: [NTOK,H] x [H,11264]
        gemm_bf16_wmma<<<dim3((2 * FFI) / BN, NTOK / BM), blk, 0, stream>>>(
            xbf, guT + (size_t)l * H * 2 * FFI, gubuf, NTOK, 2 * FFI, H);

        silu_mul_kernel<<<(NTOK * FFI) / 256, blk, 0, stream>>>(gubuf, actbf);

        // down: [NTOK,5632] x [5632,2048]
        gemm_bf16_wmma<<<dim3(H / BN, NTOK / BM), blk, 0, stream>>>(
            actbf, downT + (size_t)l * FFI * H, hbuf, NTOK, H, FFI);
    }

    // final: out = rms(h + resid) * norm_w  (fp32 to d_out)
    add_rms_kernel<<<NTOK, blk, 0, stream>>>(hbuf, resid, norm_w, nullptr, outp, 0);
}
